// SAGE_58402965291482
// MI455X (gfx1250) — compile-verified, matrix operand order checked by reading, not run on hardware
//
#include <hip/hip_runtime.h>
#include <hip/hip_bf16.h>
#include <math.h>

typedef __attribute__((ext_vector_type(16))) _Float16 v16h;
typedef __attribute__((ext_vector_type(8)))  _Float16 v8h;
typedef __attribute__((ext_vector_type(4)))  _Float16 v4h;
typedef __attribute__((ext_vector_type(8)))  float    v8f;

#define EPSN 1e-12f

// ---------------- utility kernels ----------------

__global__ void k_zero_f32(float* __restrict__ p, long long n) {
  long long i = (long long)blockIdx.x * blockDim.x + threadIdx.x;
  long long stride = (long long)gridDim.x * blockDim.x;
  for (; i < n; i += stride) p[i] = 0.0f;
}

__global__ void k_cvt_f32_f16(const float* __restrict__ in, _Float16* __restrict__ out, long long n) {
  long long i = (long long)blockIdx.x * blockDim.x + threadIdx.x;
  long long stride = (long long)gridDim.x * blockDim.x;
  for (; i < n; i += stride) out[i] = (_Float16)in[i];
}

// ---------------- graph aggregation ----------------

__global__ void k_count(const int* __restrict__ dst, float* __restrict__ cnt, int E) {
  int e = blockIdx.x * blockDim.x + threadIdx.x;
  if (e < E) unsafeAtomicAdd(&cnt[dst[e]], 1.0f);
}

// one wave (32 lanes) per edge; lane handles 4 contiguous features (8B f16 load, 4 f32 atomics)
__global__ void __launch_bounds__(256)
k_scatter(const _Float16* __restrict__ act, const int* __restrict__ src,
          const int* __restrict__ dst, float* __restrict__ sum, int E) {
  int e = blockIdx.x * 8 + (threadIdx.x >> 5);
  if (e >= E) return;
  int lane = threadIdx.x & 31;
  int s = src[e];
  int d = dst[e];
  const v4h v = *(const v4h*)(act + (size_t)s * 128 + lane * 4);
  float* o = sum + (size_t)d * 128 + lane * 4;
  unsafeAtomicAdd(o + 0, (float)v[0]);
  unsafeAtomicAdd(o + 1, (float)v[1]);
  unsafeAtomicAdd(o + 2, (float)v[2]);
  unsafeAtomicAdd(o + 3, (float)v[3]);
}

__global__ void k_divide(const float* __restrict__ sum, const float* __restrict__ cnt,
                         _Float16* __restrict__ aggH, long long n) {
  long long i = (long long)blockIdx.x * blockDim.x + threadIdx.x;
  if (i >= n) return;
  int node = (int)(i >> 7);
  aggH[i] = (_Float16)(sum[i] / fmaxf(cnt[node], 1.0f));
}

// ---------------- WMMA GEMM + L2 norm + ReLU ----------------
// 16-bit A-matrix 16x32 per-lane layout (ISA 7.12.2): lane L, m=L&15, c=L>>4,
// halfs [0..7]  = row m, K = k0 + c*8 + 0..7
// halfs [8..15] = row m, K = k0 + 16 + c*8 + 0..7
// B 32x16 uses the identical K pattern with lane = output column n, so a
// row-major weight matrix W[n][k] loads directly as the B fragment.
__device__ __forceinline__ v16h frag_ld(const _Float16* __restrict__ rowp, int k0c) {
  v8h lo = *(const v8h*)(rowp + k0c);
  v8h hi = *(const v8h*)(rowp + k0c + 16);
  return __builtin_shufflevector(lo, hi, 0, 1, 2, 3, 4, 5, 6, 7, 8, 9, 10, 11, 12, 13, 14, 15);
}

__global__ void __launch_bounds__(256)
k_gemm_norm_relu(const _Float16* __restrict__ aggH,   // [N,128] f16 mean-aggregated
                 const _Float16* __restrict__ actIn,  // [N,128] f16 layer input
                 const _Float16* __restrict__ WlH,    // [128,128] f16 row-major
                 const _Float16* __restrict__ WrH,    // [128,128] f16 row-major
                 const float* __restrict__ bias,      // [128] f32
                 _Float16* __restrict__ actOut,       // [N,128] f16 layer output
                 int nNodes) {
  __shared__ float sq[16];  // per-node sum of squares across all 8 N-tiles

  const int tid  = threadIdx.x;
  const int wave = tid >> 5;  // 0..7 -> N-tile (16 output features each)
  const int lane = tid & 31;
  const int m    = lane & 15; // A row within tile / B column within tile
  const int c    = lane >> 4;
  const int mrow = blockIdx.x * 16;

  if (tid < 16) sq[tid] = 0.0f;
  __syncthreads();

  const int rA = (mrow + m < nNodes) ? (mrow + m) : (nNodes - 1);
  const _Float16* aRow  = aggH  + (size_t)rA * 128;
  const _Float16* hRow  = actIn + (size_t)rA * 128;
  const _Float16* wlRow = WlH + (size_t)(wave * 16 + m) * 128;
  const _Float16* wrRow = WrH + (size_t)(wave * 16 + m) * 128;

  v8f acc = {};
#pragma unroll
  for (int kc = 0; kc < 4; ++kc) {
    const int k0c = kc * 32 + c * 8;
    v16h a0 = frag_ld(aRow, k0c);
    v16h b0 = frag_ld(wlRow, k0c);
    acc = __builtin_amdgcn_wmma_f32_16x16x32_f16(false, a0, false, b0, (short)0, acc, false, false);
    v16h a1 = frag_ld(hRow, k0c);
    v16h b1 = frag_ld(wrRow, k0c);
    acc = __builtin_amdgcn_wmma_f32_16x16x32_f16(false, a1, false, b1, (short)0, acc, false, false);
  }

  const float b = bias[wave * 16 + m];  // lane's output column
  float v[8];
#pragma unroll
  for (int r = 0; r < 8; ++r) {
    v[r] = acc[r] + b;                           // D element (M = r+8c, N = m)
    unsafeAtomicAdd(&sq[r + 8 * c], v[r] * v[r]);  // ds_add_f32
  }
  __syncthreads();

#pragma unroll
  for (int r = 0; r < 8; ++r) {
    const int ml = r + 8 * c;
    if (mrow + ml < nNodes) {
      const float inv = 1.0f / fmaxf(sqrtf(sq[ml]), EPSN);
      const float o = fmaxf(v[r] * inv, 0.0f);  // normalize then ReLU
      actOut[(size_t)(mrow + ml) * 128 + wave * 16 + m] = (_Float16)o;
    }
  }
}

// ---------------- final layer (d_out = 2): VALU dot + norm + log_softmax ----------------

__global__ void __launch_bounds__(256)
k_final(const _Float16* __restrict__ aggH, const _Float16* __restrict__ act,
        const float* __restrict__ Wlo, const float* __restrict__ blo,
        const float* __restrict__ Wro, float* __restrict__ out, int nNodes) {
  int i = blockIdx.x * blockDim.x + threadIdx.x;
  if (i >= nNodes) return;
  const _Float16* a = aggH + (size_t)i * 128;
  const _Float16* h = act + (size_t)i * 128;
  float acc0 = blo[0], acc1 = blo[1];
#pragma unroll 4
  for (int k = 0; k < 128; ++k) {
    const float av = (float)a[k];
    const float hv = (float)h[k];
    acc0 += av * Wlo[k] + hv * Wro[k];
    acc1 += av * Wlo[128 + k] + hv * Wro[128 + k];
  }
  const float norm = sqrtf(acc0 * acc0 + acc1 * acc1);
  const float inv = 1.0f / fmaxf(norm, EPSN);
  const float v0 = acc0 * inv, v1 = acc1 * inv;
  const float mx = fmaxf(v0, v1);
  const float lse = mx + logf(expf(v0 - mx) + expf(v1 - mx));
  out[2 * i + 0] = v0 - lse;
  out[2 * i + 1] = v1 - lse;
}

// ---------------- host launcher ----------------

extern "C" void kernel_launch(void* const* d_in, const int* in_sizes, int n_in,
                              void* d_out, int out_size, void* d_ws, size_t ws_size,
                              hipStream_t stream) {
  const float* x   = (const float*)d_in[0];
  const int*   ei  = (const int*)d_in[1];
  const float* Wl0 = (const float*)d_in[2];
  const float* bl0 = (const float*)d_in[3];
  const float* Wr0 = (const float*)d_in[4];
  const float* Wl1 = (const float*)d_in[5];
  const float* bl1 = (const float*)d_in[6];
  const float* Wr1 = (const float*)d_in[7];
  const float* Wlo = (const float*)d_in[8];
  const float* blo = (const float*)d_in[9];
  const float* Wro = (const float*)d_in[10];

  const int N = in_sizes[0] / 128;
  const int E = in_sizes[1] / 2;
  const int* src = ei;
  const int* dst = ei + E;

  // workspace carve-out (256B aligned slices)
  char* p = (char*)d_ws;
  auto carve = [&](size_t bytes) -> char* {
    char* r = p;
    p += (bytes + 255) & ~(size_t)255;
    return r;
  };
  float*    cnt  = (float*)carve((size_t)N * 4);
  float*    sum  = (float*)carve((size_t)N * 128 * 4);
  _Float16* aggH = (_Float16*)carve((size_t)N * 128 * 2);
  _Float16* actA = (_Float16*)carve((size_t)N * 128 * 2);
  _Float16* actB = (_Float16*)carve((size_t)N * 128 * 2);
  _Float16* wl0h = (_Float16*)carve((size_t)128 * 128 * 2);
  _Float16* wr0h = (_Float16*)carve((size_t)128 * 128 * 2);
  _Float16* wl1h = (_Float16*)carve((size_t)128 * 128 * 2);
  _Float16* wr1h = (_Float16*)carve((size_t)128 * 128 * 2);

  const long long ND = (long long)N * 128;
  const int B256 = 256;
  const int gND = (int)((ND + B256 - 1) / B256);
  const int gE  = (E + B256 - 1) / B256;
  const int gEw = (E + 7) / 8;   // wave-per-edge
  const int gM  = (N + 15) / 16; // node tiles for GEMM
  const int gW  = (128 * 128 + B256 - 1) / B256;

  // degree counts (shared by all layers)
  k_zero_f32<<<(N + B256 - 1) / B256, B256, 0, stream>>>(cnt, N);
  k_count<<<gE, B256, 0, stream>>>(dst, cnt, E);

  // f16 copies of x and weights
  k_cvt_f32_f16<<<gND, B256, 0, stream>>>(x, actA, ND);
  k_cvt_f32_f16<<<gW, B256, 0, stream>>>(Wl0, wl0h, 128 * 128);
  k_cvt_f32_f16<<<gW, B256, 0, stream>>>(Wr0, wr0h, 128 * 128);
  k_cvt_f32_f16<<<gW, B256, 0, stream>>>(Wl1, wl1h, 128 * 128);
  k_cvt_f32_f16<<<gW, B256, 0, stream>>>(Wr1, wr1h, 128 * 128);

  // ---- layer 0: actA -> actB ----
  k_zero_f32<<<gND, B256, 0, stream>>>(sum, ND);
  k_scatter<<<gEw, B256, 0, stream>>>(actA, src, dst, sum, E);
  k_divide<<<gND, B256, 0, stream>>>(sum, cnt, aggH, ND);
  k_gemm_norm_relu<<<gM, B256, 0, stream>>>(aggH, actA, wl0h, wr0h, bl0, actB, N);

  // ---- layer 1: actB -> actA ----
  k_zero_f32<<<gND, B256, 0, stream>>>(sum, ND);
  k_scatter<<<gEw, B256, 0, stream>>>(actB, src, dst, sum, E);
  k_divide<<<gND, B256, 0, stream>>>(sum, cnt, aggH, ND);
  k_gemm_norm_relu<<<gM, B256, 0, stream>>>(aggH, actB, wl1h, wr1h, bl1, actA, N);

  // ---- output layer: actA -> d_out [N,2] log_softmax ----
  k_zero_f32<<<gND, B256, 0, stream>>>(sum, ND);
  k_scatter<<<gEw, B256, 0, stream>>>(actA, src, dst, sum, E);
  k_divide<<<gND, B256, 0, stream>>>(sum, cnt, aggH, ND);
  k_final<<<(N + B256 - 1) / B256, B256, 0, stream>>>(aggH, actA, Wlo, blo, Wro,
                                                      (float*)d_out, N);
}
